// SingleHeadAttention_47373489275184
// MI455X (gfx1250) — matrix operand order repaired and checked by reading, not compiled, not run on hardware
//
#include <hip/hip_runtime.h>
#include <hip/hip_bf16.h>

typedef __attribute__((ext_vector_type(16))) _Float16 v16h;
typedef __attribute__((ext_vector_type(8)))  _Float16 v8h;
typedef __attribute__((ext_vector_type(8)))  float    v8f;
typedef __attribute__((ext_vector_type(4)))  int      v4i;

// Address-space-qualified pointer types for the async-to-LDS builtin:
// signature (probe-confirmed): (v4i AS1* gsrc, v4i AS3* ldst, Ii offset, Ii cpol)
typedef __attribute__((address_space(1))) v4i gv4i;
typedef __attribute__((address_space(3))) v4i lv4i;

#define WMMA_F16(A, B, C) \
  __builtin_amdgcn_wmma_f32_16x16x32_f16(false, (A), false, (B), (short)0, (C), false, false)

#if defined(__has_builtin)
# if __has_builtin(__builtin_amdgcn_global_load_async_to_lds_b128)
#  define HAS_ASYNC 1
# endif
#endif
#ifndef HAS_ASYNC
# define HAS_ASYNC 0
#endif

constexpr int D_MODEL = 1024;
constexpr int D_HEAD  = 64;
constexpr int SEQ     = 4096;
constexpr int KC      = 64;       // k-chunk staged in LDS per step (projection)
constexpr float NEG_BIG = -1.0e30f;

// ---------------------------------------------------------------------------
// Kernel 0: W [1024,64] f32 -> Wt [3][64][1024] f16 (transposed, contiguous K)
// ---------------------------------------------------------------------------
__global__ void wt_convert(const float* __restrict__ Wq,
                           const float* __restrict__ Wk,
                           const float* __restrict__ Wv,
                           _Float16* __restrict__ Wt) {
  int idx = blockIdx.x * blockDim.x + threadIdx.x;   // over 3*64*1024
  if (idx >= 3 * D_HEAD * D_MODEL) return;
  int mat = idx / (D_HEAD * D_MODEL);
  int rem = idx % (D_HEAD * D_MODEL);
  int h = rem / D_MODEL;
  int d = rem % D_MODEL;
  const float* W = (mat == 0) ? Wq : ((mat == 1) ? Wk : Wv);
  Wt[idx] = (_Float16)W[d * D_HEAD + h];   // Wt[mat][h][d]
}

// ---------------------------------------------------------------------------
// Kernel 1: QKV projection. 256-thread block = 8 waves = 8 M-tiles.
// W panels double-buffered in LDS via async-to-LDS; each wave runs
// 2 k-steps x 12 WMMAs per staged chunk, with all 12 B fragments
// preloaded so ds_loads issue as a clause instead of 12 serial waits.
//   A (16x32 f16):  lane holds row (lane&15); K = e + 16*hi  (contiguous)
//   B (32x16 f16):  lane holds col (lane&15); K = e + 16*hi  (contiguous)
//   C (16x16 f32):  lane holds col (lane&15); row = r + 8*hi
// ---------------------------------------------------------------------------
__global__ __launch_bounds__(256) void qkv_proj(const float* __restrict__ x,
                                                const _Float16* __restrict__ Wt,
                                                _Float16* __restrict__ Qh,
                                                _Float16* __restrict__ Kh,
                                                _Float16* __restrict__ Vt) {
  // [2 buffers][3 mats * 64 cols][KC halves]
  __shared__ _Float16 Wbuf[2][3 * 64 * KC] __attribute__((aligned(128)));

  const int tid   = threadIdx.x;
  const int lane  = tid & 31;
  const int wave  = tid >> 5;
  const int mtile = blockIdx.x * 8 + wave;      // 0..1023
  const int hi    = lane >> 4;
  const int ln    = lane & 15;

  const float* xr = x + (size_t)(mtile * 16 + ln) * D_MODEL;

  // Stage Wt[:, :, kc0..kc0+KC) into Wbuf[bsel]: 192 rows x 128B, 16B/op
  auto stage = [&](int bsel, int kc0) {
#pragma unroll
    for (int i = 0; i < 6; ++i) {
      int seg = tid + i * 256;              // 0..1535
      int row = seg >> 3;                   // mat*64 + col
      int s8  = seg & 7;
      const _Float16* g = Wt + (size_t)row * D_MODEL + kc0 + s8 * 8;
      _Float16* l = &Wbuf[bsel][seg * 8];
#if HAS_ASYNC
      __builtin_amdgcn_global_load_async_to_lds_b128((gv4i*)g, (lv4i*)l, 0, 0);
#else
      *(v8h*)l = *(const v8h*)g;
#endif
    }
  };

  v8f acc[3][4];
#pragma unroll
  for (int m = 0; m < 3; ++m)
#pragma unroll
    for (int n = 0; n < 4; ++n)
#pragma unroll
      for (int r = 0; r < 8; ++r) acc[m][n][r] = 0.0f;

  stage(0, 0);

  for (int c = 0; c < D_MODEL / KC; ++c) {
    const int kc0 = c * KC;
    const bool more = (c + 1 < D_MODEL / KC);
    if (more) stage((c + 1) & 1, kc0 + KC);    // overlap next panel load

#if HAS_ASYNC
    if (more) asm volatile("s_wait_asynccnt 0x6" ::: "memory");
    else      asm volatile("s_wait_asynccnt 0x0" ::: "memory");
#else
    asm volatile("s_wait_dscnt 0x0" ::: "memory");
#endif
    __syncthreads();                           // panel visible to all 8 waves

    const _Float16* wb = &Wbuf[c & 1][0];
#pragma unroll
    for (int ks = 0; ks < 2; ++ks) {
      const int k0 = kc0 + ks * 32;
      const int kl = ks * 32;
      // A fragment: two contiguous 8-float chunks of x, convert to f16
      v8f fa = *(const v8f*)(xr + k0 + hi * 16);
      v8f fb = *(const v8f*)(xr + k0 + hi * 16 + 8);
      v16h a;
#pragma unroll
      for (int e = 0; e < 8; ++e) {
        a[e]     = (_Float16)fa[e];
        a[e + 8] = (_Float16)fb[e];
      }
      // Preload all 12 B fragments (clause the ds_loads), then WMMA burst
      v16h bf[12];
#pragma unroll
      for (int m = 0; m < 3; ++m)
#pragma unroll
        for (int n = 0; n < 4; ++n)
          bf[m * 4 + n] =
              *(const v16h*)(wb + (size_t)(m * 64 + n * 16 + ln) * KC + kl + hi * 16);
#pragma unroll
      for (int m = 0; m < 3; ++m)
#pragma unroll
        for (int n = 0; n < 4; ++n)
          acc[m][n] = WMMA_F16(a, bf[m * 4 + n], acc[m][n]);
    }
    __syncthreads();                           // protect buffer before rewrite
  }

  // Store Q,K row-major f16; V transposed [B][64][S] f16
#pragma unroll
  for (int n = 0; n < 4; ++n) {
#pragma unroll
    for (int r = 0; r < 8; ++r) {
      int row = mtile * 16 + r + 8 * hi;        // global row in [0, B*S)
      int col = n * 16 + ln;                    // head dim
      Qh[(size_t)row * D_HEAD + col] = (_Float16)acc[0][n][r];
      Kh[(size_t)row * D_HEAD + col] = (_Float16)acc[1][n][r];
      int b = row >> 12;
      int s = row & (SEQ - 1);
      Vt[(size_t)b * D_HEAD * SEQ + (size_t)col * SEQ + s] = (_Float16)acc[2][n][r];
    }
  }
}

// ---------------------------------------------------------------------------
// Kernel 2: fused causal attention, online softmax, software-pipelined:
// next-K and current-V fragments prefetched while softmax VALU runs.
// One wave per (batch, 16-query tile); 32-key blocks. Tiles are remapped so
// each block holds {q, q+64, q+128, q+192} -> near-uniform per-block work.
// ---------------------------------------------------------------------------
__global__ __launch_bounds__(128) void attn(const _Float16* __restrict__ Qh,
                                            const _Float16* __restrict__ Kh,
                                            const _Float16* __restrict__ Vt,
                                            float* __restrict__ out) {
  __shared__ _Float16 Plds[4][16][32] __attribute__((aligned(64)));

  const int lane = threadIdx.x & 31;
  const int wave = threadIdx.x >> 5;
  const int gw   = blockIdx.x * 4 + wave;             // 0..1023
  const int b    = gw >> 8;                           // batch
  const int t    = ((gw & 3) << 6) | ((gw >> 2) & 63);// balanced tile remap
  const int q0   = t * 16;
  const int hi   = lane >> 4;
  const int ln   = lane & 15;

  // Q A-fragments covering h = 0..31 (aQ0) and 32..63 (aQ1)
  const _Float16* qrow = Qh + (size_t)(b * SEQ + q0 + ln) * D_HEAD;
  v16h aQ0, aQ1;
  {
    v8h c0 = *(const v8h*)(qrow + hi * 16);
    v8h c1 = *(const v8h*)(qrow + hi * 16 + 8);
    v8h c2 = *(const v8h*)(qrow + 32 + hi * 16);
    v8h c3 = *(const v8h*)(qrow + 40 + hi * 16);
#pragma unroll
    for (int e = 0; e < 8; ++e) {
      aQ0[e] = c0[e]; aQ0[e + 8] = c1[e];
      aQ1[e] = c2[e]; aQ1[e + 8] = c3[e];
    }
  }

  const _Float16* Kbase = Kh + (size_t)b * SEQ * D_HEAD;
  const _Float16* Vbase = Vt + (size_t)b * D_HEAD * SEQ;

  auto loadK = [&](int k0, v16h& b0a, v16h& b0b, v16h& b1a, v16h& b1b) {
    const _Float16* kr0 = Kbase + (size_t)(k0 + ln) * D_HEAD;
    const _Float16* kr1 = Kbase + (size_t)(k0 + 16 + ln) * D_HEAD;
    b0a = *(const v16h*)(kr0 + hi * 16);        // h 0..31
    b0b = *(const v16h*)(kr0 + 32 + hi * 16);   // h 32..63
    b1a = *(const v16h*)(kr1 + hi * 16);
    b1b = *(const v16h*)(kr1 + 32 + hi * 16);
  };

  float mi[8], li[8];
  v8f accO[4];
#pragma unroll
  for (int r = 0; r < 8; ++r) { mi[r] = NEG_BIG; li[r] = 0.0f; }
#pragma unroll
  for (int n = 0; n < 4; ++n)
#pragma unroll
    for (int r = 0; r < 8; ++r) accO[n][r] = 0.0f;

  const int kbmax = (q0 + 15) >> 5;
  v16h kc0a, kc0b, kc1a, kc1b;                 // current block K fragments
  loadK(0, kc0a, kc0b, kc1a, kc1b);

  for (int kb = 0; kb <= kbmax; ++kb) {
    const int k0 = kb * 32;

    // ---- S = Q K^T for 32 keys --------------------------------------------
    v8f s[2];
#pragma unroll
    for (int r = 0; r < 8; ++r) { s[0][r] = 0.0f; s[1][r] = 0.0f; }
    s[0] = WMMA_F16(aQ0, kc0a, s[0]);
    s[0] = WMMA_F16(aQ1, kc0b, s[0]);
    s[1] = WMMA_F16(aQ0, kc1a, s[1]);
    s[1] = WMMA_F16(aQ1, kc1b, s[1]);

    // ---- prefetch: V for this block, K for next block (hide under softmax) -
    v16h bv[4];
#pragma unroll
    for (int n = 0; n < 4; ++n) {
      const int h = n * 16 + ln;
      bv[n] = *(const v16h*)(Vbase + (size_t)h * SEQ + k0 + 16 * hi);
    }
    v16h kn0a, kn0b, kn1a, kn1b;
    if (kb < kbmax) loadK(k0 + 32, kn0a, kn0b, kn1a, kn1b);

    // ---- causal mask + online softmax --------------------------------------
    const int key0 = k0 + ln;
    const int key1 = k0 + 16 + ln;
    float alpha[8];
#pragma unroll
    for (int r = 0; r < 8; ++r) {
      const int q = q0 + r + 8 * hi;
      float s0 = (key0 <= q) ? s[0][r] * 0.125f : NEG_BIG;
      float s1 = (key1 <= q) ? s[1][r] * 0.125f : NEG_BIG;
      float mx = fmaxf(s0, s1);
#pragma unroll
      for (int m2 = 8; m2 >= 1; m2 >>= 1) mx = fmaxf(mx, __shfl_xor(mx, m2, 16));
      float mn = fmaxf(mi[r], mx);
      float a  = __expf(mi[r] - mn);
      float p0 = __expf(s0 - mn);
      float p1 = __expf(s1 - mn);
      float rs = p0 + p1;
#pragma unroll
      for (int m2 = 8; m2 >= 1; m2 >>= 1) rs += __shfl_xor(rs, m2, 16);
      li[r] = a * li[r] + rs;
      mi[r] = mn;
      alpha[r] = a;
      // Stage P (C-layout) into LDS row-major [16][32]
      Plds[wave][r + 8 * hi][ln]      = (_Float16)p0;
      Plds[wave][r + 8 * hi][16 + ln] = (_Float16)p1;
    }

    // Rescale O accumulators (row mapping matches stats: row = r + 8*hi)
#pragma unroll
    for (int n = 0; n < 4; ++n)
#pragma unroll
      for (int r = 0; r < 8; ++r) accO[n][r] *= alpha[r];

    // Cross-lane LDS dependency within one wave
    asm volatile("s_wait_dscnt 0x0" ::: "memory");

    // Reload P as A-fragment (row = ln; K = e + 16*hi)
    v8h pa = *(const v8h*)(&Plds[wave][ln][hi * 16]);
    v8h pb = *(const v8h*)(&Plds[wave][ln][hi * 16 + 8]);
    v16h aP;
#pragma unroll
    for (int e = 0; e < 8; ++e) { aP[e] = pa[e]; aP[e + 8] = pb[e]; }

    // ---- O += P * V (V fragments already in registers) ---------------------
#pragma unroll
    for (int n = 0; n < 4; ++n) accO[n] = WMMA_F16(aP, bv[n], accO[n]);

    kc0a = kn0a; kc0b = kn0b; kc1a = kn1a; kc1b = kn1b;
  }

  // ---- epilogue: normalize and store f32 -----------------------------------
#pragma unroll
  for (int n = 0; n < 4; ++n) {
#pragma unroll
    for (int r = 0; r < 8; ++r) {
      const int row = q0 + r + 8 * hi;
      out[((size_t)b * SEQ + row) * D_HEAD + n * 16 + ln] = accO[n][r] / li[r];
    }
  }
}

// ---------------------------------------------------------------------------
extern "C" void kernel_launch(void* const* d_in, const int* in_sizes, int n_in,
                              void* d_out, int out_size, void* d_ws, size_t ws_size,
                              hipStream_t stream) {
  const float* x  = (const float*)d_in[0];
  const float* Wq = (const float*)d_in[1];
  const float* Wk = (const float*)d_in[2];
  const float* Wv = (const float*)d_in[3];
  float* out = (float*)d_out;

  char* ws = (char*)d_ws;
  // Workspace layout (bytes):
  //   Wt : 3*64*1024*2   = 0x060000
  //   Qh : 16384*64*2    = 0x200000
  //   Kh : 16384*64*2    = 0x200000
  //   Vt : 4*64*4096*2   = 0x200000   -> total 0x660000 (~6.7 MB)
  _Float16* Wt = (_Float16*)(ws);
  _Float16* Qh = (_Float16*)(ws + 0x060000);
  _Float16* Kh = (_Float16*)(ws + 0x260000);
  _Float16* Vt = (_Float16*)(ws + 0x460000);

  wt_convert<<<768, 256, 0, stream>>>(Wq, Wk, Wv, Wt);
  qkv_proj<<<128, 256, 0, stream>>>(x, Wt, Qh, Kh, Vt);
  attn<<<256, 128, 0, stream>>>(Qh, Kh, Vt, out);
}
